// BinaryLinear_61529701482733
// MI455X (gfx1250) — compile-verified, matrix operand order checked by reading, not run on hardware
//
#include <hip/hip_runtime.h>

typedef __attribute__((ext_vector_type(16))) _Float16 v16h;
typedef __attribute__((ext_vector_type(8)))  _Float16 v8h;
typedef __attribute__((ext_vector_type(4)))  _Float16 v4h;
typedef __attribute__((ext_vector_type(4)))  unsigned short v4u16;
typedef __attribute__((ext_vector_type(8)))  float    v8f;
typedef __attribute__((ext_vector_type(4)))  float    v4f;
typedef __attribute__((ext_vector_type(4)))  int      v4i;

#define BM 128
#define BN 128
#define BK 32
#define PITCH 40   // halfs per LDS row: 32 data + 8 pad (80B rows, 16B aligned)

// ---------------- CDNA5 async global->LDS helpers ----------------
#if __has_builtin(__builtin_amdgcn_global_load_async_to_lds_b128)
#define HAS_ASYNC_LDS 1
#endif

typedef __attribute__((address_space(1))) v4i* gv4i_p;
typedef __attribute__((address_space(3))) v4i* lv4i_p;

__device__ __forceinline__ void async_copy_b128(const void* g, void* l) {
#ifdef HAS_ASYNC_LDS
    __builtin_amdgcn_global_load_async_to_lds_b128(
        (gv4i_p)(g), (lv4i_p)(l), /*offset=*/0, /*cpol=*/0);
#else
    *(v8h*)l = *(const v8h*)g;   // synchronous fallback, same layout
#endif
}

__device__ __forceinline__ void wait_async_all() {
#ifdef HAS_ASYNC_LDS
#if __has_builtin(__builtin_amdgcn_s_wait_asynccnt)
    __builtin_amdgcn_s_wait_asynccnt(0);
#else
    asm volatile("s_wait_asynccnt 0x0" ::: "memory");
#endif
#endif
}

// sign(w) as f16 bits: ±1.0h keeps W's sign bit; exact 0 -> 0 (torch.sign)
__device__ __forceinline__ unsigned short sign_f16_bits(float w) {
    unsigned int u = __float_as_uint(w);
    unsigned short h = (unsigned short)(0x3C00u | ((u >> 16) & 0x8000u));
    return (w != 0.f) ? h : (unsigned short)0;
}

// ---------------- pre-pass: fp32 -> f16 (x) ----------------
__global__ __launch_bounds__(256)
void cvt_x_f16(const float* __restrict__ in, _Float16* __restrict__ out, long n)
{
    long i = ((long)blockIdx.x * blockDim.x + threadIdx.x) * 4;
    const long stride = (long)gridDim.x * blockDim.x * 4;
    for (; i < n; i += stride) {
        v4f v = *(const v4f*)(in + i);
        v4h h;
        #pragma unroll
        for (int q = 0; q < 4; ++q) h[q] = (_Float16)v[q];
        *(v4h*)(out + i) = h;
    }
}

// ---------------- pre-pass: fp32 -> sign as f16 (W) ----------------
__global__ __launch_bounds__(256)
void cvt_w_sign_f16(const float* __restrict__ in, _Float16* __restrict__ out, long n)
{
    long i = ((long)blockIdx.x * blockDim.x + threadIdx.x) * 4;
    const long stride = (long)gridDim.x * blockDim.x * 4;
    for (; i < n; i += stride) {
        v4f v = *(const v4f*)(in + i);
        v4u16 h;
        #pragma unroll
        for (int q = 0; q < 4; ++q) h[q] = sign_f16_bits(v[q]);
        *(v4u16*)(out + i) = h;
    }
}

// ---------------- main GEMM: f16 tiles via async-to-LDS + WMMA ----------------
__global__ __launch_bounds__(256)
void gemm_f16_wmma(const _Float16* __restrict__ Xh,  // [M,K] f16
                   const _Float16* __restrict__ Wh,  // [N,K] f16 (already sign)
                   const float* __restrict__ Bv,     // [N]
                   float* __restrict__ Out,          // [M,N]
                   int M, int N, int K)
{
    __shared__ __align__(16) _Float16 As[2][BM * PITCH];
    __shared__ __align__(16) _Float16 Bs[2][BN * PITCH];

    const int tid  = threadIdx.x;
    const int lane = tid & 31;
    const int wave = tid >> 5;      // 0..7
    const int wr   = wave >> 1;     // 0..3 : 32-row M band
    const int wc   = wave & 1;      // 0..1 : 64-col N band

    const long m0 = (long)blockIdx.y * BM;
    const long n0 = (long)blockIdx.x * BN;

    // async loader mapping: 16B per lane; tile row = 64B -> 4 lanes/row
    const int arow = tid >> 2;        // 0..63
    const int acol = (tid & 3) * 8;   // half offset: 0,8,16,24

    v8f acc[2][4] = {};
    const int nk = K / BK;

    auto issue_tile = [&](int kt, int buf) {
        const _Float16* xs = Xh + (m0 + arow) * (long)K + (long)kt * BK + acol;
        const _Float16* ws = Wh + (n0 + arow) * (long)K + (long)kt * BK + acol;
        _Float16* la = &As[buf][arow * PITCH + acol];
        _Float16* lb = &Bs[buf][arow * PITCH + acol];
        async_copy_b128(xs,             la);
        async_copy_b128(xs + 64L * K,   la + 64 * PITCH);
        async_copy_b128(ws,             lb);
        async_copy_b128(ws + 64L * K,   lb + 64 * PITCH);
    };

    const int rsel = lane & 15;
    const int ksel = (lane >> 4) * 8;   // ISA 16-bit A/B fragment layout

    auto compute = [&](int buf) {
        v16h afr[2], bfr[4];
        #pragma unroll
        for (int i = 0; i < 2; ++i) {
            const _Float16* p = &As[buf][(wr * 32 + i * 16 + rsel) * PITCH];
            ((v8h*)&afr[i])[0] = *(const v8h*)(p + ksel);
            ((v8h*)&afr[i])[1] = *(const v8h*)(p + ksel + 16);
        }
        #pragma unroll
        for (int j = 0; j < 4; ++j) {
            const _Float16* p = &Bs[buf][(wc * 64 + j * 16 + rsel) * PITCH];
            ((v8h*)&bfr[j])[0] = *(const v8h*)(p + ksel);
            ((v8h*)&bfr[j])[1] = *(const v8h*)(p + ksel + 16);
        }
        #pragma unroll
        for (int i = 0; i < 2; ++i)
            #pragma unroll
            for (int j = 0; j < 4; ++j)
                acc[i][j] = __builtin_amdgcn_wmma_f32_16x16x32_f16(
                    false, afr[i], false, bfr[j], (short)0, acc[i][j], false, false);
    };

    issue_tile(0, 0);
    wait_async_all();
    __syncthreads();

    for (int kt = 0; kt < nk; ++kt) {
        const int buf = kt & 1;
        if (kt + 1 < nk) issue_tile(kt + 1, buf ^ 1);  // async: overlaps compute
        compute(buf);
        wait_async_all();                              // tile kt+1 resident in LDS
        __syncthreads();
    }

    #pragma unroll
    for (int j = 0; j < 4; ++j) {
        const long n = n0 + wc * 64 + j * 16 + (lane & 15);
        const float bias = Bv[n];
        #pragma unroll
        for (int i = 0; i < 2; ++i) {
            const long mbase = m0 + wr * 32 + i * 16 + (lane >> 4) * 8;
            #pragma unroll
            for (int r = 0; r < 8; ++r)
                Out[(mbase + r) * N + n] = acc[i][j][r] + bias;
        }
    }
}

// ---------------- fallback: fused fp32 kernel (ws too small) ----------------
__global__ __launch_bounds__(256)
void binlin_fused(const float* __restrict__ X, const float* __restrict__ W,
                  const float* __restrict__ Bv, float* __restrict__ Out,
                  int M, int N, int K)
{
    __shared__ __align__(16) _Float16 As[2][BM * PITCH];
    __shared__ __align__(16) _Float16 Bs[2][BN * PITCH];

    const int tid  = threadIdx.x;
    const int lane = tid & 31;
    const int wave = tid >> 5;
    const int wr   = wave >> 1;
    const int wc   = wave & 1;

    const long m0 = (long)blockIdx.y * BM;
    const long n0 = (long)blockIdx.x * BN;

    const int lrow = tid >> 3;
    const int lcol = (tid & 7) * 4;

    v8f acc[2][4] = {};
    const int nk = K / BK;
    v4f ax[4], bw[4];

    auto gload = [&](int kt) {
        const float* xp = X + m0 * K + (long)kt * BK;
        const float* wp = W + n0 * K + (long)kt * BK;
        #pragma unroll
        for (int p = 0; p < 4; ++p) {
            long r = lrow + p * 32;
            ax[p] = *(const v4f*)(xp + r * K + lcol);
            bw[p] = *(const v4f*)(wp + r * K + lcol);
        }
    };
    auto sstore = [&](int buf) {
        #pragma unroll
        for (int p = 0; p < 4; ++p) {
            int r = lrow + p * 32;
            v4h ha; v4u16 hb;
            #pragma unroll
            for (int q = 0; q < 4; ++q) {
                ha[q] = (_Float16)ax[p][q];
                hb[q] = sign_f16_bits(bw[p][q]);
            }
            *(v4h*)&As[buf][r * PITCH + lcol] = ha;
            *(v4u16*)&Bs[buf][r * PITCH + lcol] = hb;
        }
    };

    const int rsel = lane & 15;
    const int ksel = (lane >> 4) * 8;

    auto compute = [&](int buf) {
        v16h afr[2], bfr[4];
        #pragma unroll
        for (int i = 0; i < 2; ++i) {
            const _Float16* p = &As[buf][(wr * 32 + i * 16 + rsel) * PITCH];
            ((v8h*)&afr[i])[0] = *(const v8h*)(p + ksel);
            ((v8h*)&afr[i])[1] = *(const v8h*)(p + ksel + 16);
        }
        #pragma unroll
        for (int j = 0; j < 4; ++j) {
            const _Float16* p = &Bs[buf][(wc * 64 + j * 16 + rsel) * PITCH];
            ((v8h*)&bfr[j])[0] = *(const v8h*)(p + ksel);
            ((v8h*)&bfr[j])[1] = *(const v8h*)(p + ksel + 16);
        }
        #pragma unroll
        for (int i = 0; i < 2; ++i)
            #pragma unroll
            for (int j = 0; j < 4; ++j)
                acc[i][j] = __builtin_amdgcn_wmma_f32_16x16x32_f16(
                    false, afr[i], false, bfr[j], (short)0, acc[i][j], false, false);
    };

    gload(0); sstore(0);
    __syncthreads();
    if (nk > 1) gload(1);

    for (int kt = 0; kt < nk; ++kt) {
        const int buf = kt & 1;
        if (kt + 1 < nk) sstore(buf ^ 1);
        if (kt + 2 < nk) gload(kt + 2);
        compute(buf);
        __syncthreads();
    }

    #pragma unroll
    for (int j = 0; j < 4; ++j) {
        const long n = n0 + wc * 64 + j * 16 + (lane & 15);
        const float bias = Bv[n];
        #pragma unroll
        for (int i = 0; i < 2; ++i) {
            const long mbase = m0 + wr * 32 + i * 16 + (lane >> 4) * 8;
            #pragma unroll
            for (int r = 0; r < 8; ++r)
                Out[(mbase + r) * N + n] = acc[i][j][r] + bias;
        }
    }
}

extern "C" void kernel_launch(void* const* d_in, const int* in_sizes, int n_in,
                              void* d_out, int out_size, void* d_ws, size_t ws_size,
                              hipStream_t stream)
{
    const float* x  = (const float*)d_in[0];
    const float* w  = (const float*)d_in[1];
    const float* bv = (const float*)d_in[2];
    float* out = (float*)d_out;

    const int D = in_sizes[2];         // 4096
    const int M = in_sizes[0] / D;     // B*S
    const int N = D, K = D;

    const size_t nx = (size_t)M * K;
    const size_t nw = (size_t)N * K;
    const size_t need = (nx + nw) * sizeof(_Float16);

    dim3 grid(N / BN, M / BM);

    if (ws_size >= need) {
        _Float16* xh = (_Float16*)d_ws;
        _Float16* wh = xh + nx;
        int cb = (int)((nx / 4 + 255) / 256);
        if (cb > 65535) cb = 65535;
        cvt_x_f16     <<<cb, 256, 0, stream>>>(x, xh, (long)nx);
        cvt_w_sign_f16<<<cb, 256, 0, stream>>>(w, wh, (long)nw);
        gemm_f16_wmma <<<grid, 256, 0, stream>>>(xh, wh, bv, out, M, N, K);
    } else {
        binlin_fused  <<<grid, 256, 0, stream>>>(x, w, bv, out, M, N, K);
    }
}
